// SemanticConsistencyGNN_11553462026404
// MI455X (gfx1250) — matrix operand order repaired and checked by reading, not compiled, not run on hardware
//
#include <hip/hip_runtime.h>
#include <hip/hip_bf16.h>

#define N_NODES 468
#define BATCH   128
#define MROWS   (BATCH * N_NODES)   // 59904, multiple of 32

typedef __attribute__((ext_vector_type(16))) __bf16 v16bf;
typedef __attribute__((ext_vector_type(8)))  __bf16 v8bf;
typedef __attribute__((ext_vector_type(8)))  float  v8f;

__device__ __forceinline__ __bf16 f2bf(float f) {
  unsigned u = __builtin_bit_cast(unsigned, f);
  unsigned r = (u + 0x7FFFu + ((u >> 16) & 1u)) >> 16;   // round-to-nearest-even
  return __builtin_bit_cast(__bf16, (unsigned short)r);
}

__device__ __forceinline__ v16bf cat16(v8bf lo, v8bf hi) {
  return __builtin_shufflevector(lo, hi, 0, 1, 2, 3, 4, 5, 6, 7,
                                         8, 9, 10, 11, 12, 13, 14, 15);
}

// ---- prep: W [K,Nd] f32 -> Wt [Nd,K] bf16 (row n = column n of W, contiguous in K)
__global__ void k_wt_bf16(const float* __restrict__ W, __bf16* __restrict__ Wt,
                          int K, int Nd) {
  int idx = blockIdx.x * blockDim.x + threadIdx.x;
  if (idx >= K * Nd) return;
  int n = idx / K, k = idx - n * K;
  Wt[idx] = f2bf(W[k * Nd + n]);
}

// ---- prep: banded normalized-adjacency coefficients coef[i][o+9] = dinv(i)*dinv(i+o)
__global__ void k_coef(float* __restrict__ coef) {
  int idx = blockIdx.x * blockDim.x + threadIdx.x;
  if (idx >= N_NODES * 19) return;
  int i = idx / 19, o = (idx - i * 19) - 9;
  int j = i + o;
  float v = 0.0f;
  if (j >= 0 && j < N_NODES) {
    float di = 1.0f + (float)min(9, N_NODES - 1 - i) + (float)min(9, i);
    float dj = 1.0f + (float)min(9, N_NODES - 1 - j) + (float)min(9, j);
    v = rsqrtf(di) * rsqrtf(dj);
  }
  coef[idx] = v;
}

// ---- node encoder: [M,3] @ [3,64] + b  (K=3, plain VALU), writes bf16 activations
__global__ void k_encoder(const float* __restrict__ lm, const float* __restrict__ W,
                          const float* __restrict__ b, __bf16* __restrict__ X) {
  int idx = blockIdx.x * blockDim.x + threadIdx.x;
  if (idx >= MROWS * 64) return;
  int c = idx & 63, row = idx >> 6;
  float a0 = lm[row * 3 + 0], a1 = lm[row * 3 + 1], a2 = lm[row * 3 + 2];
  float acc = b[c];
  acc = fmaf(a0, W[c], acc);
  acc = fmaf(a1, W[64 + c], acc);
  acc = fmaf(a2, W[128 + c], acc);
  X[idx] = f2bf(acc);
}

// ---- WMMA GEMM: H[M,Nd] = A[M,K](bf16) @ Wt[Nd,K]^T(bf16), f32 accum.
// Each wave owns a 32x64 output block (2 M-tiles x 4 N-tiles, 8 accumulators).
// Per k-step: 6 fragment loads (12 b128) feed 8 back-to-back WMMAs.
// K multiple of 32; M multiple of 32; Nd multiple of 64.
__global__ __launch_bounds__(256) void k_gemm_wmma(
    const __bf16* __restrict__ A, const __bf16* __restrict__ Wt,
    const float* __restrict__ bias, float* __restrict__ H,
    int M, int K, int Nd, int fuse_relu) {
  int wave    = blockIdx.x * 8 + (threadIdx.x >> 5);
  int tilesNw = Nd >> 6;                      // groups of four 16-wide N tiles
  int tiles   = (M >> 5) * tilesNw;
  if (wave >= tiles) return;                  // wave-uniform: EXEC stays all-ones
  int tm = wave / tilesNw, tw = wave - tm * tilesNw;
  int lane = threadIdx.x & 31;
  int half = lane >> 4, mr = lane & 15;

  const __bf16* arow0 = A + (size_t)(tm * 32 + mr) * K;        // M rows 0..15
  const __bf16* arow1 = arow0 + (size_t)16 * K;                // M rows 16..31
  const __bf16* wbas  = Wt + (size_t)(tw * 64 + mr) * K;

  v8f acc[2][4] = {{{}, {}, {}, {}}, {{}, {}, {}, {}}};
  for (int k0 = 0; k0 < K; k0 += 32) {
    // A fragments: K = 8*half + {0..7} and 8*half + {16..23}, 16B-aligned loads
    int ka = k0 + half * 8;
    v16bf a0 = cat16(*(const v8bf*)(arow0 + ka), *(const v8bf*)(arow0 + ka + 16));
    v16bf a1 = cat16(*(const v8bf*)(arow1 + ka), *(const v8bf*)(arow1 + ka + 16));
    // B fragments for columns (tw*64 + nt*16 + mr): K = 8*half + {0..7} and 8*half + {16..23}
    v16bf bfr[4];
    int kb = k0 + half * 8;
#pragma unroll
    for (int nt = 0; nt < 4; ++nt) {
      const __bf16* wr = wbas + (size_t)(nt * 16) * K + kb;
      bfr[nt] = cat16(*(const v8bf*)wr, *(const v8bf*)(wr + 16));
    }
#pragma unroll
    for (int nt = 0; nt < 4; ++nt) {
      acc[0][nt] = __builtin_amdgcn_wmma_f32_16x16x32_bf16(
          false, a0, false, bfr[nt], (short)0, acc[0][nt], false, false);
      acc[1][nt] = __builtin_amdgcn_wmma_f32_16x16x32_bf16(
          false, a1, false, bfr[nt], (short)0, acc[1][nt], false, false);
    }
  }

#pragma unroll
  for (int mt = 0; mt < 2; ++mt) {
#pragma unroll
    for (int nt = 0; nt < 4; ++nt) {
      int col = tw * 64 + nt * 16 + mr;
      float bv = bias ? bias[col] : 0.0f;
#pragma unroll
      for (int r = 0; r < 8; ++r) {           // D layout: VGPR r -> M = r + 8*half
        float v = acc[mt][nt][r] + bv;
        if (fuse_relu) v = fmaxf(v, 0.0f);
        H[(size_t)(tm * 32 + mt * 16 + r + half * 8) * Nd + col] = v;
      }
    }
  }
}

// ---- banded aggregation (19-tap stencil along node dim) + bias + ReLU, f32 -> bf16
__global__ void k_agg(const float* __restrict__ H, const float* __restrict__ coef,
                      const float* __restrict__ bias, __bf16* __restrict__ Y, int D) {
  int idx = blockIdx.x * blockDim.x + threadIdx.x;
  int total = BATCH * N_NODES * D;
  if (idx >= total) return;
  int c   = idx % D;
  int rem = idx / D;
  int i   = rem % N_NODES;
  int b   = rem / N_NODES;
  const float* hb = H + (size_t)b * N_NODES * D + c;
  const float* cf = coef + i * 19;
  float acc = bias[c];
#pragma unroll
  for (int o = -9; o <= 9; ++o) {
    int j = i + o;
    if (j >= 0 && j < N_NODES)
      acc = fmaf(cf[o + 9], hb[(size_t)j * D], acc);
  }
  Y[idx] = f2bf(fmaxf(acc, 0.0f));
}

// ---- global mean pool over nodes: bf16 [B,N,128] -> f32 [B,128] (+ bf16 copy)
__global__ void k_pool(const __bf16* __restrict__ X, float* __restrict__ G,
                       __bf16* __restrict__ Gb) {
  int idx = blockIdx.x * blockDim.x + threadIdx.x;
  if (idx >= BATCH * 128) return;
  int c = idx & 127, b = idx >> 7;
  const __bf16* x = X + (size_t)b * N_NODES * 128 + c;
  float s = 0.0f;
  for (int n = 0; n < N_NODES; ++n) s += (float)x[(size_t)n * 128];
  float m = s * (1.0f / (float)N_NODES);
  G[idx]  = m;
  Gb[idx] = f2bf(m);
}

// ---- classifier: [B,512] @ [512,2] + b
__global__ void k_cls(const float* __restrict__ F, const float* __restrict__ W,
                      const float* __restrict__ b, float* __restrict__ O) {
  int idx = blockIdx.x * blockDim.x + threadIdx.x;
  if (idx >= BATCH * 2) return;
  int c = idx & 1, bb = idx >> 1;
  const float* f = F + (size_t)bb * 512;
  float acc = b[c];
  for (int k = 0; k < 512; ++k) acc = fmaf(f[k], W[k * 2 + c], acc);
  O[idx] = acc;
}

extern "C" void kernel_launch(void* const* d_in, const int* in_sizes, int n_in,
                              void* d_out, int out_size, void* d_ws, size_t ws_size,
                              hipStream_t stream) {
  const float* lm    = (const float*)d_in[0];
  const float* W_enc = (const float*)d_in[1];
  const float* b_enc = (const float*)d_in[2];
  const float* W1    = (const float*)d_in[3];
  const float* b1    = (const float*)d_in[4];
  const float* W2    = (const float*)d_in[5];
  const float* b2    = (const float*)d_in[6];
  const float* W3    = (const float*)d_in[7];
  const float* b3    = (const float*)d_in[8];
  const float* W_fus = (const float*)d_in[9];
  const float* b_fus = (const float*)d_in[10];
  const float* W_cls = (const float*)d_in[11];
  const float* b_cls = (const float*)d_in[12];

  // d_out = [output 128*2][features 128*512][graph_features 128*128]
  float* out_ptr  = (float*)d_out;
  float* feat_ptr = out_ptr + BATCH * 2;
  float* gfeat    = feat_ptr + BATCH * 512;

  // workspace carve-out: bf16 X (30.7MB) + f32 H (61.3MB) + coef + bf16 weights
  char*  ws  = (char*)d_ws;
  size_t off = 0;
  auto carve = [&](size_t bytes) -> void* {
    void* p = ws + off;
    off = (off + bytes + 255) & ~(size_t)255;
    return p;
  };
  __bf16* Xb    = (__bf16*)carve((size_t)MROWS * 256 * sizeof(__bf16));
  float*  Hf    = (float*)carve((size_t)MROWS * 256 * sizeof(float));
  float*  coef  = (float*)carve((size_t)N_NODES * 19 * sizeof(float));
  __bf16* Wt1   = (__bf16*)carve((size_t)128 * 64  * sizeof(__bf16));
  __bf16* Wt2   = (__bf16*)carve((size_t)256 * 128 * sizeof(__bf16));
  __bf16* Wt3   = (__bf16*)carve((size_t)128 * 256 * sizeof(__bf16));
  __bf16* Wtf   = (__bf16*)carve((size_t)512 * 128 * sizeof(__bf16));
  __bf16* Gb    = (__bf16*)carve((size_t)BATCH * 128 * sizeof(__bf16));
  (void)ws_size; (void)in_sizes; (void)n_in; (void)out_size;

  const int T = 256;
  // prep
  k_wt_bf16<<<(64 * 128 + T - 1) / T,  T, 0, stream>>>(W1,    Wt1, 64,  128);
  k_wt_bf16<<<(128 * 256 + T - 1) / T, T, 0, stream>>>(W2,    Wt2, 128, 256);
  k_wt_bf16<<<(256 * 128 + T - 1) / T, T, 0, stream>>>(W3,    Wt3, 256, 128);
  k_wt_bf16<<<(128 * 512 + T - 1) / T, T, 0, stream>>>(W_fus, Wtf, 128, 512);
  k_coef<<<(N_NODES * 19 + T - 1) / T, T, 0, stream>>>(coef);

  // encoder -> Xb [M,64] bf16
  k_encoder<<<(MROWS * 64 + T - 1) / T, T, 0, stream>>>(lm, W_enc, b_enc, Xb);

  // layer 1: gemm Xb[M,64] -> Hf[M,128]; agg Hf -> Xb (bf16)
  {
    int tiles = (MROWS / 32) * (128 / 64);
    k_gemm_wmma<<<(tiles + 7) / 8, T, 0, stream>>>(Xb, Wt1, nullptr, Hf, MROWS, 64, 128, 0);
    int tot = BATCH * N_NODES * 128;
    k_agg<<<(tot + T - 1) / T, T, 0, stream>>>(Hf, coef, b1, Xb, 128);
  }
  // layer 2: gemm Xb[M,128] -> Hf[M,256]; agg Hf -> Xb
  {
    int tiles = (MROWS / 32) * (256 / 64);
    k_gemm_wmma<<<(tiles + 7) / 8, T, 0, stream>>>(Xb, Wt2, nullptr, Hf, MROWS, 128, 256, 0);
    int tot = BATCH * N_NODES * 256;
    k_agg<<<(tot + T - 1) / T, T, 0, stream>>>(Hf, coef, b2, Xb, 256);
  }
  // layer 3: gemm Xb[M,256] -> Hf[M,128]; agg Hf -> Xb
  {
    int tiles = (MROWS / 32) * (128 / 64);
    k_gemm_wmma<<<(tiles + 7) / 8, T, 0, stream>>>(Xb, Wt3, nullptr, Hf, MROWS, 256, 128, 0);
    int tot = BATCH * N_NODES * 128;
    k_agg<<<(tot + T - 1) / T, T, 0, stream>>>(Hf, coef, b3, Xb, 128);
  }
  // mean pool -> graph_features (f32, d_out) + bf16 copy for fusion GEMM
  k_pool<<<(BATCH * 128 + T - 1) / T, T, 0, stream>>>(Xb, gfeat, Gb);
  // fusion: [128,128] @ [128,512] + b, ReLU -> features (d_out slice)
  {
    int tiles = (BATCH / 32) * (512 / 64);
    k_gemm_wmma<<<(tiles + 7) / 8, T, 0, stream>>>(Gb, Wtf, b_fus, feat_ptr, BATCH, 128, 512, 1);
  }
  // classifier -> output (d_out slice)
  k_cls<<<1, T, 0, stream>>>(feat_ptr, W_cls, b_cls, out_ptr);
}
